// SimplifiedGateElmanCell_47768626266222
// MI455X (gfx1250) — matrix operand order repaired and checked by reading, not compiled, not run on hardware
//
#include <hip/hip_runtime.h>
#include <hip/hip_bf16.h>
#include <math.h>

// ---------------------------------------------------------------------------
// SimplifiedGateElmanCell for MI455X (gfx1250, wave32, WMMA)
//   T=2048, B=32, D=1024
//   out[t] = h[t+1] * silu(Wx[t] + h[t+1] + b_gate),  h[t+1]=tanh(Wx[t]+h[t]Wh_n^T+b)
// d_out = [ output (T*B*D f32) | h (T+1)*B*D f32 ]
// Wx_all is staged into the `output` region and overwritten in-place by the
// recurrence epilogue (saves a 256MB workspace).
// ---------------------------------------------------------------------------

typedef __attribute__((ext_vector_type(16))) __bf16 v16bf;
typedef __attribute__((ext_vector_type(8)))  float  v8f;
typedef __attribute__((ext_vector_type(4)))  int    v4i;

union Frag16 {           // one WMMA bf16 operand fragment (8 VGPRs / lane)
    v16bf v;
    uint4 q[2];
    __bf16 e[16];
};

#define TT 2048
#define BB 32
#define DD 1024

// Async load-to-LDS path (CDNA5 GLOBAL_LOAD_ASYNC_TO_LDS_B128 + ASYNCcnt),
// guarded so a toolchain without the builtins falls back to sync staging.
// Probe-learned signature: (v4i addrspace(1)*, v4i addrspace(3)*, Imm, Imm).
#if defined(__has_builtin)
# if __has_builtin(__builtin_amdgcn_global_load_async_to_lds_b128) && \
     __has_builtin(__builtin_amdgcn_s_wait_asynccnt)
#  define USE_ASYNC_LDS 1
# endif
#endif

#if defined(USE_ASYNC_LDS)
typedef __attribute__((address_space(1))) v4i* as1_v4i;
typedef __attribute__((address_space(3))) v4i* as3_v4i;
#define ASYNC_CP16(g, l, off) \
    __builtin_amdgcn_global_load_async_to_lds_b128((as1_v4i)(g), (as3_v4i)(l), (off), 0)
#endif

// ---------------------------------------------------------------------------
// Block-wide sum reduction (wave32 shuffles + LDS), result broadcast to all.
// ---------------------------------------------------------------------------
__device__ __forceinline__ float block_reduce_sum(float x, float* red) {
    #pragma unroll
    for (int o = 16; o > 0; o >>= 1) x += __shfl_xor(x, o);
    const int w = threadIdx.x >> 5;
    if ((threadIdx.x & 31) == 0) red[w] = x;
    __syncthreads();
    const int nw = blockDim.x >> 5;
    float s = (threadIdx.x < nw) ? red[threadIdx.x] : 0.0f;
    if (w == 0) {
        #pragma unroll
        for (int o = 16; o > 0; o >>= 1) s += __shfl_xor(s, o);
        if (threadIdx.x == 0) red[0] = s;
    }
    __syncthreads();
    const float r = red[0];
    __syncthreads();
    return r;
}

// ---------------------------------------------------------------------------
// Kernel 1: 3-step power iteration -> scale = 0.99/(sigma+1e-8). One WG.
// ---------------------------------------------------------------------------
__global__ __launch_bounds__(1024) void spectral_kernel(
        const float* __restrict__ Wh, const float* __restrict__ u0,
        float* __restrict__ scaleOut) {
    __shared__ float u[DD];
    __shared__ float v[DD];
    __shared__ float red[32];
    const int tid = threadIdx.x;

    float uv = u0[tid];
    float n0 = block_reduce_sum(uv * uv, red);
    u[tid] = uv / sqrtf(n0);                 // u = u0/||u0|| (no eps, per ref)
    __syncthreads();

    float wv = 0.0f;                         // keeps last (W @ v) component
    #pragma unroll 1
    for (int it = 0; it < 3; ++it) {
        // v = W^T u   (coalesced: fixed i -> contiguous across tid)
        float a = 0.0f;
        #pragma unroll 4
        for (int i = 0; i < DD; ++i) a += Wh[i * DD + tid] * u[i];
        float nv = sqrtf(block_reduce_sum(a * a, red)) + 1e-8f;
        v[tid] = a / nv;
        __syncthreads();
        // u = W v
        float b = 0.0f;
        #pragma unroll 4
        for (int k = 0; k < DD; ++k) b += Wh[tid * DD + k] * v[k];
        wv = b;
        float nu = sqrtf(block_reduce_sum(b * b, red)) + 1e-8f;
        u[tid] = b / nu;
        __syncthreads();
    }
    // sigma = |u . (W v_last)|
    float sig = block_reduce_sum(u[tid] * wv, red);
    if (tid == 0) scaleOut[0] = 0.99f / (fabsf(sig) + 1e-8f);
}

// ---------------------------------------------------------------------------
// Helpers: f32x8 -> bf16x8 conversion (lowers to v_cvt_pk_bf16_f32)
// ---------------------------------------------------------------------------
__device__ __forceinline__ void cvt8(__bf16* d, float4 a, float4 b) {
    d[0] = (__bf16)a.x; d[1] = (__bf16)a.y; d[2] = (__bf16)a.z; d[3] = (__bf16)a.w;
    d[4] = (__bf16)b.x; d[5] = (__bf16)b.y; d[6] = (__bf16)b.z; d[7] = (__bf16)b.w;
}

// ---------------------------------------------------------------------------
// Kernel 2: Wx_all = x @ Wx^T  (M=T*B=65536, N=K=1024), bf16 WMMA, f32 accum.
// WG tile 128x64, 8 waves (4x2), wave tile 32x32 -> 4 C tiles of 16x16.
// Async path: double-buffered f32 tiles DMA'd to LDS (ASYNCcnt), bf16
// conversion fused into the fragment-read path.
// ---------------------------------------------------------------------------
__global__ __launch_bounds__(256) void wx_gemm_kernel(
        const float* __restrict__ X,          // [M, 1024] f32
        const float* __restrict__ Wxm,        // [1024, 1024] f32 (row e, col d)
        float* __restrict__ Out) {            // [M, 1024] f32
    const int tid  = threadIdx.x;
    const int lane = tid & 31;
    const int wave = tid >> 5;                // 0..7
    const int wm   = wave >> 1;               // 0..3 -> 32 rows each
    const int wn   = wave & 1;                // 0..1 -> 32 cols each
    const int lm   = lane & 15;
    const bool hi  = lane >= 16;

    const int row0 = blockIdx.y * 128;
    const int n0   = blockIdx.x * 64;

    v8f acc[2][2] = {{{}, {}}, {{}, {}}};

#if defined(USE_ASYNC_LDS)
    __shared__ float Af[2][128 * 32];         // 2 x 16KB A tiles (f32)
    __shared__ float Bf[2][64 * 32];          // 2 x  8KB Bt tiles (f32)

    // Per-thread DMA slices: A -> 4 x b128, B -> 2 x b128 (6 issues/wave/tile)
    const int arow = tid >> 1;                // 0..127
    const int acg  = (tid & 1) * 16;          // 16 floats
    const int brow = tid >> 2;                // 0..63
    const int bcg  = (tid & 3) * 8;           // 8 floats
    const float* gA = X   + (size_t)(row0 + arow) * DD + acg;
    const float* gB = Wxm + (size_t)(n0  + brow) * DD + bcg;

    #define ISSUE_TILE(c)                                                     \
        do {                                                                  \
            const int kb_ = (c) * 32, bf_ = (c) & 1;                          \
            const float* ga_ = gA + kb_;                                      \
            float* la_ = &Af[bf_][arow * 32 + acg];                           \
            ASYNC_CP16(ga_, la_, 0);  ASYNC_CP16(ga_, la_, 16);               \
            ASYNC_CP16(ga_, la_, 32); ASYNC_CP16(ga_, la_, 48);               \
            const float* gb_ = gB + kb_;                                      \
            float* lb_ = &Bf[bf_][brow * 32 + bcg];                           \
            ASYNC_CP16(gb_, lb_, 0);  ASYNC_CP16(gb_, lb_, 16);               \
        } while (0)

    ISSUE_TILE(0);

    #pragma unroll 1
    for (int c = 0; c < 32; ++c) {
        const int cur = c & 1;
        if (c < 31) {                          // prefetch next tile into other buf
            ISSUE_TILE(c + 1);
            __builtin_amdgcn_s_wait_asynccnt(6);   // batch c retired, c+1 in flight
        } else {
            __builtin_amdgcn_s_wait_asynccnt(0);
        }
        __syncthreads();                       // whole tile visible to all waves

        // B fragments once per tile (reused across mt)
        Frag16 bfr[2];
        #pragma unroll
        for (int nt = 0; nt < 2; ++nt) {
            const int bc   = wn * 32 + nt * 16 + lm;
            const int koff = hi ? 8 : 0;
            const float* bp = &Bf[cur][bc * 32 + koff];
            float4 f0 = ((const float4*)bp)[0];
            float4 f1 = ((const float4*)bp)[1];
            float4 f2 = ((const float4*)(bp + 16))[0];
            float4 f3 = ((const float4*)(bp + 16))[1];
            cvt8(&bfr[nt].e[0], f0, f1);
            cvt8(&bfr[nt].e[8], f2, f3);
        }
        #pragma unroll
        for (int mt = 0; mt < 2; ++mt) {
            Frag16 a;
            const int ar   = wm * 32 + mt * 16 + lm;
            const int aoff = hi ? 8 : 0;
            const float* ap = &Af[cur][ar * 32 + aoff];
            float4 f0 = ((const float4*)ap)[0];
            float4 f1 = ((const float4*)ap)[1];
            float4 f2 = ((const float4*)(ap + 16))[0];
            float4 f3 = ((const float4*)(ap + 16))[1];
            cvt8(&a.e[0], f0, f1);
            cvt8(&a.e[8], f2, f3);
            #pragma unroll
            for (int nt = 0; nt < 2; ++nt)
                acc[mt][nt] = __builtin_amdgcn_wmma_f32_16x16x32_bf16(
                    false, a.v, false, bfr[nt].v, (short)0, acc[mt][nt], false, false);
        }
        __syncthreads();                       // done reading buf before reuse
    }
    #undef ISSUE_TILE
#else
    // -------- synchronous fallback: VGPR-staged bf16 tiles in LDS ----------
    __shared__ __bf16 Abuf[128 * 32];
    __shared__ __bf16 Bbuf[64 * 32];
    const int arow = tid >> 1;
    const int acg  = (tid & 1) * 16;
    const int brow = (tid & 127) >> 1;
    const int bcg  = (tid & 1) * 16;

    #pragma unroll 1
    for (int kb = 0; kb < DD; kb += 32) {
        __syncthreads();
        {
            const float* s = X + (size_t)(row0 + arow) * DD + kb + acg;
            float4 f0 = ((const float4*)s)[0];
            float4 f1 = ((const float4*)s)[1];
            float4 f2 = ((const float4*)s)[2];
            float4 f3 = ((const float4*)s)[3];
            if (kb + 32 < DD) __builtin_prefetch(s + 32, 0, 1);
            __bf16 tmp[16];
            cvt8(tmp, f0, f1); cvt8(tmp + 8, f2, f3);
            __bf16* d = &Abuf[arow * 32 + acg];
            *(uint4*)d       = *(uint4*)&tmp[0];
            *(uint4*)(d + 8) = *(uint4*)&tmp[8];
        }
        if (tid < 128) {
            const float* s = Wxm + (size_t)(n0 + brow) * DD + kb + bcg;
            float4 f0 = ((const float4*)s)[0];
            float4 f1 = ((const float4*)s)[1];
            float4 f2 = ((const float4*)s)[2];
            float4 f3 = ((const float4*)s)[3];
            __bf16 tmp[16];
            cvt8(tmp, f0, f1); cvt8(tmp + 8, f2, f3);
            __bf16* d = &Bbuf[brow * 32 + bcg];
            *(uint4*)d       = *(uint4*)&tmp[0];
            *(uint4*)(d + 8) = *(uint4*)&tmp[8];
        }
        __syncthreads();

        #pragma unroll
        for (int mt = 0; mt < 2; ++mt) {
            Frag16 a;
            const int ar   = wm * 32 + mt * 16 + lm;
            const int aoff = hi ? 8 : 0;
            a.q[0] = *(const uint4*)&Abuf[ar * 32 + aoff];
            a.q[1] = *(const uint4*)&Abuf[ar * 32 + aoff + 16];
            #pragma unroll
            for (int nt = 0; nt < 2; ++nt) {
                Frag16 b;
                const int bc   = wn * 32 + nt * 16 + lm;
                const int boff = hi ? 8 : 0;
                b.q[0] = *(const uint4*)&Bbuf[bc * 32 + boff];
                b.q[1] = *(const uint4*)&Bbuf[bc * 32 + boff + 16];
                acc[mt][nt] = __builtin_amdgcn_wmma_f32_16x16x32_bf16(
                    false, a.v, false, b.v, (short)0, acc[mt][nt], false, false);
            }
        }
    }
#endif

    // epilogue: C layout -> VGPR r holds M=r (lanes 0-15) / M=r+8 (lanes 16-31)
    #pragma unroll
    for (int mt = 0; mt < 2; ++mt)
        #pragma unroll
        for (int nt = 0; nt < 2; ++nt)
            #pragma unroll
            for (int r = 0; r < 8; ++r) {
                const int row = row0 + wm * 32 + mt * 16 + r + (hi ? 8 : 0);
                const int col = n0 + wn * 32 + nt * 16 + lm;
                Out[(size_t)row * DD + col] = acc[mt][nt][r];
            }
}

// ---------------------------------------------------------------------------
// Device-wide two-phase barrier (persistent kernel). __threadfence supplies
// the agent-scope writeback/invalidate so h[t] is visible across WGPs.
// ---------------------------------------------------------------------------
__device__ __forceinline__ void grid_sync(unsigned* cnt, unsigned* gen,
                                          unsigned nb, unsigned* my) {
    __threadfence();
    __syncthreads();
    if (threadIdx.x == 0) {
        const unsigned g = *my;
        if (atomicAdd(cnt, 1u) == nb - 1u) {
            atomicExch(cnt, 0u);
            __threadfence();
            atomicAdd(gen, 1u);
        } else {
            while (atomicAdd(gen, 0u) == g) __builtin_amdgcn_s_sleep(2);
        }
    }
    __syncthreads();
    __threadfence();
    *my += 1u;
}

// ---------------------------------------------------------------------------
// Kernel 3: persistent recurrence. 64 WGs x 64 threads (2 waves).
// WG g owns columns e in [16g,16g+16); wave m in {0,1} owns rows 16m..16m+15.
// W_h*scale -> bf16 staged ONCE into LDS; per step: 32 wmma (full K=1024),
// fused tanh/silu epilogue, grid barrier.
// ---------------------------------------------------------------------------
__global__ __launch_bounds__(64) void recur_kernel(
        const float* __restrict__ Wh,         // raw W_h [D,D]
        const float* __restrict__ bvec,
        const float* __restrict__ bgate,
        float* __restrict__ outbuf,           // [T,B,D]: holds Wx on entry
        float* __restrict__ hbuf,             // [T+1,B,D]: h[0]=h0 on entry
        const float* __restrict__ wsScale,
        unsigned* __restrict__ barCnt,
        unsigned* __restrict__ barGen) {
    __shared__ __bf16 Bs[16 * DD];            // Bs[n][k] = W_h_n[n0+n][k], 32KB

    const int tid  = threadIdx.x;
    const int lane = tid & 31;
    const int m    = tid >> 5;                // wave -> row half
    const int lm   = lane & 15;
    const bool hi  = lane >= 16;
    const int n0   = blockIdx.x * 16;

    const float scale = *wsScale;
    for (int idx = tid; idx < 16 * DD; idx += 64) {
        const int n = idx >> 10, k = idx & (DD - 1);
        Bs[n * DD + k] = (__bf16)(Wh[(size_t)(n0 + n) * DD + k] * scale);
    }
    const float bc = bvec[n0 + lm];
    const float bg = bgate[n0 + lm];
    __syncthreads();

    unsigned gen = 0;
    #pragma unroll 1
    for (int t = 0; t < TT; ++t) {
        const float* hp = hbuf + (size_t)t * BB * DD;
        v8f acc = {};
        #pragma unroll 4
        for (int ks = 0; ks < 32; ++ks) {
            const int kb = ks * 32;
            // A fragment from h[t] (f32 -> bf16 in registers)
            Frag16 a;
            const int aoff = hi ? 8 : 0;
            const float* ap = hp + (size_t)(m * 16 + lm) * DD + kb + aoff;
            float4 f0 = ((const float4*)ap)[0];
            float4 f1 = ((const float4*)ap)[1];
            float4 f2 = ((const float4*)(ap + 16))[0];
            float4 f3 = ((const float4*)(ap + 16))[1];
            cvt8(&a.e[0], f0, f1);
            cvt8(&a.e[8], f2, f3);
            // B fragment from LDS-resident W_h_n slice
            Frag16 b;
            const __bf16* bp = &Bs[lm * DD + kb + (hi ? 8 : 0)];
            b.q[0] = *(const uint4*)bp;
            b.q[1] = *(const uint4*)(bp + 16);
            acc = __builtin_amdgcn_wmma_f32_16x16x32_bf16(
                false, a.v, false, b.v, (short)0, acc, false, false);
        }
        // fused epilogue: raw->tanh, gate->silu, write out + h[t+1]
        #pragma unroll
        for (int r = 0; r < 8; ++r) {
            const int row = m * 16 + r + (hi ? 8 : 0);
            const size_t oidx = ((size_t)t * BB + row) * DD + n0 + lm;
            const float wx  = outbuf[oidx];
            const float raw = wx + acc[r] + bc;
            const float hn  = tanhf(raw);
            const float pre = wx + hn + bg;
            const float sig = 1.0f / (1.0f + __expf(-pre));
            outbuf[oidx] = hn * (pre * sig);
            hbuf[(((size_t)t + 1) * BB + row) * DD + n0 + lm] = hn;
        }
        grid_sync(barCnt, barGen, gridDim.x, &gen);
    }
}

// ---------------------------------------------------------------------------
extern "C" void kernel_launch(void* const* d_in, const int* in_sizes, int n_in,
                              void* d_out, int out_size, void* d_ws, size_t ws_size,
                              hipStream_t stream) {
    (void)in_sizes; (void)n_in; (void)out_size; (void)ws_size;
    const float* x  = (const float*)d_in[0];
    // d_in[1] = z (unused by gate_mode 0)
    const float* h0 = (const float*)d_in[2];
    const float* Wx = (const float*)d_in[3];
    const float* Wh = (const float*)d_in[4];
    const float* b  = (const float*)d_in[5];
    const float* bg = (const float*)d_in[6];
    const float* u0 = (const float*)d_in[7];

    float* outbuf = (float*)d_out;                       // [T,B,D]
    float* hbuf   = outbuf + (size_t)TT * BB * DD;       // [T+1,B,D]

    unsigned* bar   = (unsigned*)d_ws;                   // [0]=count,[1]=gen
    float*    scale = (float*)((char*)d_ws + 64);

    hipMemsetAsync(d_ws, 0, 128, stream);                             // barrier reset (deterministic)
    hipMemcpyAsync(hbuf, h0, (size_t)BB * DD * sizeof(float),
                   hipMemcpyDeviceToDevice, stream);                  // h[0] = h0

    spectral_kernel<<<1, 1024, 0, stream>>>(Wh, u0, scale);
    wx_gemm_kernel<<<dim3(16, 512), 256, 0, stream>>>(x, Wx, outbuf); // Wx_all -> out region
    recur_kernel<<<64, 64, 0, stream>>>(Wh, b, bg, outbuf, hbuf, scale,
                                        bar, bar + 1);
}
